// Model_39676907887708
// MI455X (gfx1250) — compile-verified, matrix operand order checked by reading, not run on hardware
//
#include <hip/hip_runtime.h>

#define B_SZ 4
#define LQ   2048
#define LKV  2048
#define DIM  1024
#define LSHIFT 11              // log2(LQ) == log2(LKV)
#define LMASK  (LKV - 1)

typedef __attribute__((ext_vector_type(16))) __bf16 bf16x16;
typedef __attribute__((ext_vector_type(8)))  float  f32x8;

struct __attribute__((aligned(16))) U128 { unsigned int a[4]; };
struct __attribute__((aligned(16))) F128 { float f[4]; };

union Frag {
    bf16x16        v;
    U128           q[2];
    unsigned short s[16];
};

__device__ __forceinline__ unsigned short f32_to_bf16(float x) {
    unsigned int u = __float_as_uint(x);
    u += 0x7FFFu + ((u >> 16) & 1u);          // round-to-nearest-even
    return (unsigned short)(u >> 16);
}

// Load a 16x32 bf16 row-major tile (rows x K) into an A/B-style WMMA fragment.
// Per ISA: lane<16 -> row=lane, K=[0..7],[16..23]; lane>=16 -> row=lane-16, K=[8..15],[24..31].
__device__ __forceinline__ void load_frag_b16(Frag& f, const unsigned short* base, int stride) {
    const int lane = threadIdx.x & 31;
    const int r    = lane & 15;
    const int kb   = (lane < 16) ? 0 : 8;
    const unsigned short* p = base + (size_t)r * stride + kb;
    f.q[0] = *reinterpret_cast<const U128*>(p);
    f.q[1] = *reinterpret_cast<const U128*>(p + 16);
}

__device__ __forceinline__ f32x8 wmma_bf16(const Frag& a, const Frag& b, f32x8 c) {
    return __builtin_amdgcn_wmma_f32_16x16x32_bf16(false, a.v, false, b.v, (short)0, c, false, false);
}

// CDNA5 async copy: 16 bytes global -> LDS, tracked on ASYNCcnt (no VGPR staging).
__device__ __forceinline__ void async_copy_b128(const unsigned short* gptr, const unsigned short* lptr) {
    unsigned int lds = (unsigned int)(uintptr_t)lptr;   // low 32 bits of generic LDS ptr = LDS byte addr
    unsigned long long ga = (unsigned long long)(uintptr_t)gptr;
    asm volatile("global_load_async_to_lds_b128 %0, %1, off" :: "v"(lds), "v"(ga) : "memory");
}
// async loads complete in order: waiting <=8 releases the 8 oldest (current tile)
__device__ __forceinline__ void wait_async_le8() { asm volatile("s_wait_asynccnt 0x8" ::: "memory"); }
__device__ __forceinline__ void wait_async_all() { asm volatile("s_wait_asynccnt 0x0" ::: "memory"); }

// ---------------------------------------------------------------------------
// Elementwise f32 -> bf16 (one-time pre-convert so GEMMs move pure bf16)
// ---------------------------------------------------------------------------
__global__ __launch_bounds__(256) void cvt_f32_bf16(const float* __restrict__ src,
                                                    unsigned short* __restrict__ dst, int n) {
    int i = (blockIdx.x * 256 + threadIdx.x) * 8;
    if (i >= n) return;
    const F128* p = reinterpret_cast<const F128*>(src + i);
    F128 v0 = p[0], v1 = p[1];
    union { U128 u; unsigned short s[8]; } o;
#pragma unroll
    for (int e = 0; e < 4; ++e) o.s[e]     = f32_to_bf16(v0.f[e]);
#pragma unroll
    for (int e = 0; e < 4; ++e) o.s[4 + e] = f32_to_bf16(v1.f[e]);
    *reinterpret_cast<U128*>(dst + i) = o.u;
}

// ---------------------------------------------------------------------------
// Projection GEMM (bf16 in, bf16 out): Out[m,n] = sum_k X[m,k] * W[n,k].
// transpose_out=1 stores Out as [B, N, L] (for V^T) with m = b*L + t, L=2048.
// Block tile 128x128, K-chunk 64, double-buffered LDS filled with
// GLOBAL_LOAD_ASYNC_TO_LDS_B128 (next tile streams in under the WMMAs).
// ---------------------------------------------------------------------------
#define TROW 72                 // 64 cols padded to 72 ushorts (144B rows, 16B-aligned)
#define TSZ  (128 * TROW)

__global__ __launch_bounds__(256) void proj_gemm_bf16(
    const unsigned short* __restrict__ X, const unsigned short* __restrict__ W,
    unsigned short* __restrict__ Out, int N, int K, int transpose_out)
{
    __shared__ unsigned short Xs[2 * TSZ];
    __shared__ unsigned short Ws[2 * TSZ];

    const int tid  = threadIdx.x;
    const int bm   = blockIdx.x, bn = blockIdx.y;
    const int w    = tid >> 5, lane = tid & 31;
    const int mw   = w & 3, nw = w >> 2;            // 4 waves in M, 2 in N
    const int ncol = lane & 15, hi = lane >> 4;

    f32x8 acc[2][4];
#pragma unroll
    for (int i = 0; i < 2; ++i)
#pragma unroll
        for (int j = 0; j < 4; ++j) acc[i][j] = {};

    const int lrow  = tid >> 1;                     // 0..127
    const int lcolh = (tid & 1) * 32;               // 0 or 32 (each thread copies 64B/matrix)
    const unsigned short* xrow = X + (size_t)(bm * 128 + lrow) * K + lcolh;
    const unsigned short* wrow = W + (size_t)(bn * 128 + lrow) * K + lcolh;
    unsigned short* xl = Xs + lrow * TROW + lcolh;
    unsigned short* wl = Ws + lrow * TROW + lcolh;

    // prologue: stream first tile into buffer 0
#pragma unroll
    for (int c = 0; c < 4; ++c) {
        async_copy_b128(xrow + c * 8, xl + c * 8);
        async_copy_b128(wrow + c * 8, wl + c * 8);
    }

    int cur = 0;
    for (int k0 = 0; k0 < K; k0 += 64) {
        if (k0 + 64 < K) {
            const int nxt = cur ^ 1;
#pragma unroll
            for (int c = 0; c < 4; ++c) {           // issue next tile, then wait only for current
                async_copy_b128(xrow + k0 + 64 + c * 8, xl + nxt * TSZ + c * 8);
                async_copy_b128(wrow + k0 + 64 + c * 8, wl + nxt * TSZ + c * 8);
            }
            wait_async_le8();
        } else {
            wait_async_all();
        }
        __syncthreads();

        const unsigned short* xb = Xs + cur * TSZ;
        const unsigned short* wb = Ws + cur * TSZ;
#pragma unroll
        for (int ks = 0; ks < 2; ++ks) {
            Frag a[2], bq[4];
#pragma unroll
            for (int i = 0; i < 2; ++i) load_frag_b16(a[i],  xb + (32 * mw + 16 * i) * TROW + 32 * ks, TROW);
#pragma unroll
            for (int j = 0; j < 4; ++j) load_frag_b16(bq[j], wb + (64 * nw + 16 * j) * TROW + 32 * ks, TROW);
#pragma unroll
            for (int i = 0; i < 2; ++i)
#pragma unroll
                for (int j = 0; j < 4; ++j)
                    acc[i][j] = wmma_bf16(a[i], bq[j], acc[i][j]);
        }
        __syncthreads();    // all waves done with buf `cur` before it is refilled next+1 iter
        cur ^= 1;
    }

#pragma unroll
    for (int i = 0; i < 2; ++i)
#pragma unroll
        for (int j = 0; j < 4; ++j)
#pragma unroll
            for (int v = 0; v < 8; ++v) {
                int gm = bm * 128 + 32 * mw + 16 * i + v + 8 * hi;
                int gn = bn * 128 + 64 * nw + 16 * j + ncol;
                unsigned short h = f32_to_bf16(acc[i][j][v]);
                if (!transpose_out) {
                    Out[(size_t)gm * N + gn] = h;
                } else {
                    int bb = gm >> LSHIFT;          // L == 2048: shift/mask, no divide
                    int tt = gm & LMASK;
                    Out[((size_t)bb * N << LSHIFT) + ((size_t)gn << LSHIFT) + tt] = h;
                }
            }
}

// ---------------------------------------------------------------------------
// Flash attention: one block = 16 queries of one batch. 8 waves, each owns a
// 128-wide D-slice; O accumulator (16x128 f32/wave) in registers. 64 keys per
// step; partial QK^T reduced across waves via ds_add_f32; softmax writes a
// bf16 P tile to LDS which feeds the PV WMMAs via ds_load_b128.
// ---------------------------------------------------------------------------
__global__ __launch_bounds__(256) void flash_attn_bf16(
    const unsigned short* __restrict__ Q, const unsigned short* __restrict__ Km,
    const unsigned short* __restrict__ Vt, float* __restrict__ Out)
{
    __shared__ float S[16][68];                 // raw score accumulation tile
    __shared__ unsigned short Pb[16][72];       // bf16 softmax tile (144B rows, 16B-aligned)
    __shared__ float red[16][17];
    __shared__ float m_run[16], l_run[16], alf[16];

    const int tid  = threadIdx.x;
    const int w    = tid >> 5, lane = tid & 31;
    const int ncol = lane & 15, hi = lane >> 4;
    const int q0   = blockIdx.x * 16;
    const int b    = blockIdx.y;
    const int d0   = w * 128;
    const float inv_scale = 1.0f / 32.0f;       // 1/sqrt(1024)

    Frag qf[4];
    const unsigned short* qbase = Q + ((size_t)(b * LQ + q0)) * DIM + d0;
#pragma unroll
    for (int kk = 0; kk < 4; ++kk) load_frag_b16(qf[kk], qbase + 32 * kk, DIM);

    f32x8 oacc[8];
#pragma unroll
    for (int n = 0; n < 8; ++n) oacc[n] = {};

    if (tid < 16) { m_run[tid] = -__builtin_inff(); l_run[tid] = 0.0f; }
    __syncthreads();

    const int r   = tid >> 4;
    const int c16 = tid & 15;

    for (int t0 = 0; t0 < LKV; t0 += 64) {
        // prefetch next K block for this wave's D-slice (global_prefetch_b8)
        if (t0 + 64 < LKV) {
            const unsigned short* nk = Km + ((size_t)(b * LKV + t0 + 64 + (lane & 31))) * DIM + d0;
            __builtin_prefetch(nk, 0, 1);
            __builtin_prefetch(nk + 64, 0, 1);
        }

        for (int i = tid; i < 16 * 68; i += 256) (&S[0][0])[i] = 0.0f;
        __syncthreads();

        f32x8 sacc[4];
#pragma unroll
        for (int nt = 0; nt < 4; ++nt) sacc[nt] = {};
#pragma unroll
        for (int nt = 0; nt < 4; ++nt)
#pragma unroll
            for (int kk = 0; kk < 4; ++kk) {
                Frag kf;
                load_frag_b16(kf, Km + ((size_t)(b * LKV + t0 + 16 * nt)) * DIM + d0 + 32 * kk, DIM);
                sacc[nt] = wmma_bf16(qf[kk], kf, sacc[nt]);
            }
#pragma unroll
        for (int nt = 0; nt < 4; ++nt)
#pragma unroll
            for (int v = 0; v < 8; ++v)
                atomicAdd(&S[v + 8 * hi][16 * nt + ncol], sacc[nt][v]);
        __syncthreads();

        // ---- online softmax over 16x64 tile (16 threads per row) ----
        float vals[4], mx = -__builtin_inff();
#pragma unroll
        for (int j = 0; j < 4; ++j) {
            vals[j] = S[r][c16 + 16 * j] * inv_scale;
            mx = fmaxf(mx, vals[j]);
        }
        red[r][c16] = mx;
        __syncthreads();
        float rowmax = -__builtin_inff();
#pragma unroll
        for (int k = 0; k < 16; ++k) rowmax = fmaxf(rowmax, red[r][k]);
        const float mold = m_run[r];
        const float mnew = fmaxf(mold, rowmax);
        float psum = 0.0f;
#pragma unroll
        for (int j = 0; j < 4; ++j) {
            float p = __expf(vals[j] - mnew);
            Pb[r][c16 + 16 * j] = f32_to_bf16(p);   // bf16 P tile, ready for WMMA
            psum += p;
        }
        __syncthreads();
        red[r][c16] = psum;
        __syncthreads();
        if (c16 == 0) {
            float s = 0.0f;
#pragma unroll
            for (int k = 0; k < 16; ++k) s += red[r][k];
            float alpha = __expf(mold - mnew);
            l_run[r] = alpha * l_run[r] + s;
            m_run[r] = mnew;
            alf[r]   = alpha;
        }
        __syncthreads();

        float av[8];
#pragma unroll
        for (int v = 0; v < 8; ++v) av[v] = alf[v + 8 * hi];
#pragma unroll
        for (int n = 0; n < 8; ++n)
#pragma unroll
            for (int v = 0; v < 8; ++v) oacc[n][v] *= av[v];

        Frag pf[2];
#pragma unroll
        for (int kc = 0; kc < 2; ++kc) load_frag_b16(pf[kc], &Pb[0][0] + 32 * kc, 72);
#pragma unroll
        for (int n = 0; n < 8; ++n)
#pragma unroll
            for (int kc = 0; kc < 2; ++kc) {
                Frag vf;
                load_frag_b16(vf, Vt + ((size_t)b * DIM + d0 + 16 * n) * LKV + t0 + 32 * kc, LKV);
                oacc[n] = wmma_bf16(pf[kc], vf, oacc[n]);
            }
        __syncthreads();
    }

    float linv[8];
#pragma unroll
    for (int v = 0; v < 8; ++v) linv[v] = 1.0f / l_run[v + 8 * hi];
#pragma unroll
    for (int n = 0; n < 8; ++n)
#pragma unroll
        for (int v = 0; v < 8; ++v)
            Out[((size_t)(b * LQ + q0 + v + 8 * hi)) * DIM + d0 + 16 * n + ncol] =
                oacc[n][v] * linv[v];
}

extern "C" void kernel_launch(void* const* d_in, const int* in_sizes, int n_in,
                              void* d_out, int out_size, void* d_ws, size_t ws_size,
                              hipStream_t stream) {
    (void)in_sizes; (void)n_in; (void)out_size; (void)ws_size;
    const float* x1  = (const float*)d_in[0];
    const float* x2  = (const float*)d_in[1];
    const float* w_q = (const float*)d_in[2];
    const float* w_k = (const float*)d_in[3];
    const float* w_v = (const float*)d_in[4];
    float* out = (float*)d_out;

    const size_t NX = (size_t)B_SZ * LQ * DIM;   // 8M elems
    const size_t NW = (size_t)DIM * DIM;         // 1M elems

    unsigned short* p = (unsigned short*)d_ws;
    unsigned short *Qws = p,           *Kws = Qws + NX,  *Vtws = Kws + NX;
    unsigned short *x1b = Vtws + NX,   *x2b = x1b + NX;
    unsigned short *wqb = x2b + NX,    *wkb = wqb + NW,  *wvb = wkb + NW;

    cvt_f32_bf16<<<dim3((unsigned)(NX / 8 / 256)), 256, 0, stream>>>(x1, x1b, (int)NX);
    cvt_f32_bf16<<<dim3((unsigned)(NX / 8 / 256)), 256, 0, stream>>>(x2, x2b, (int)NX);
    cvt_f32_bf16<<<dim3((unsigned)(NW / 8 / 256)), 256, 0, stream>>>(w_q, wqb, (int)NW);
    cvt_f32_bf16<<<dim3((unsigned)(NW / 8 / 256)), 256, 0, stream>>>(w_k, wkb, (int)NW);
    cvt_f32_bf16<<<dim3((unsigned)(NW / 8 / 256)), 256, 0, stream>>>(w_v, wvb, (int)NW);

    dim3 pgrid(B_SZ * LQ / 128, DIM / 128);
    proj_gemm_bf16<<<pgrid, 256, 0, stream>>>(x1b, wqb, Qws,  DIM, DIM, 0);
    proj_gemm_bf16<<<pgrid, 256, 0, stream>>>(x2b, wkb, Kws,  DIM, DIM, 0);
    proj_gemm_bf16<<<pgrid, 256, 0, stream>>>(x2b, wvb, Vtws, DIM, DIM, 1);

    dim3 agrid(LQ / 16, B_SZ);
    flash_attn_bf16<<<agrid, 256, 0, stream>>>(Qws, Kws, Vtws, out);
}
